// VASNet_Audio128_Attention_31765578121237
// MI455X (gfx1250) — matrix operand order repaired and checked by reading, not compiled, not run on hardware
//
#include <hip/hip_runtime.h>
#include <hip/hip_bf16.h>

// ---------------------------------------------------------------------------
// VASNet audio-visual attention, CDNA5 (gfx1250, wave32).
//   - Dense GEMMs: v_wmma_f32_16x16x32_f16, f32 accumulate, half operands
//     pre-converted in separate bandwidth-bound passes.  One wave computes a
//     32x64 tile (8 WMMA / K-step, 2x A-frag + 4x B-frag reuse), K-loop
//     software-pipelined (next fragments loaded before current WMMAs issue).
//   - Banded softmax / att.T-apply on VALU (0.7 GFLOP, memory-bound).
//   - Outputs: out[4096] then att_weights[4096*4096] fp32, concatenated.
// ---------------------------------------------------------------------------

typedef __attribute__((ext_vector_type(16))) _Float16 v16h;
typedef __attribute__((ext_vector_type(8)))  _Float16 v8h;
typedef __attribute__((ext_vector_type(4)))  _Float16 v4h;
typedef __attribute__((ext_vector_type(8)))  float    v8f;

#define N_SEQ 4096
#define M_DIM 1024
#define A_DIM 128

static __device__ __forceinline__ float wave_sum(float v) {
#pragma unroll
  for (int m = 16; m > 0; m >>= 1) v += __shfl_xor(v, m, 32);
  return v;
}
static __device__ __forceinline__ float wave_max(float v) {
#pragma unroll
  for (int m = 16; m > 0; m >>= 1) v = fmaxf(v, __shfl_xor(v, m, 32));
  return v;
}

// Two aligned 16B half-vectors -> one 16-element WMMA fragment.
static __device__ __forceinline__ v16h ld16(const _Float16* __restrict__ p0,
                                            const _Float16* __restrict__ p1) {
  v8h lo = *(const v8h*)p0;
  v8h hi = *(const v8h*)p1;
  return __builtin_shufflevector(lo, hi, 0, 1, 2, 3, 4, 5, 6, 7,
                                 8, 9, 10, 11, 12, 13, 14, 15);
}

// ------------------------------ GEMM kernel --------------------------------
// C[M,N] = A[M,K] @ B[N,K]^T (+bias[N]) (optional relu).  A,B are f16
// (row-major, K contiguous).  One wave per 32x64 tile: per K-step of 32,
// 2 A-fragments x 4 B-fragments feed 8 WMMAs; next step's fragments are
// loaded before the current WMMAs so loads overlap the matrix pipe.
//
// A-frag (16x32): lane&15 = M row; lanes<16 hold K 0..15,
// lanes>=16 hold K 16..31 (2 halves per VGPR).
// B-frag (32x16): lane&15 = N col; lanes<16 hold K 0..15, lanes>=16 K 16..31.
__global__ void wmma_gemm_h(const _Float16* __restrict__ A, const _Float16* __restrict__ B,
                            const float* __restrict__ bias, float* __restrict__ C,
                            int Mrows, int Nout, int K, int relu) {
  const int lane = threadIdx.x & 31;
  const int wave = threadIdx.x >> 5;
  const int tilesN = Nout >> 6;            // 64-wide N tiles
  const int gt = blockIdx.x * (blockDim.x >> 5) + wave;
  const int tm = gt / tilesN;              // 32-high M tiles
  const int tn = gt % tilesN;
  if (tm >= (Mrows >> 5)) return;          // wave-uniform guard

  const int l15 = lane & 15;
  const int hiA = (lane >= 16) ? 16 : 0;   // half-index offset within K-step
  const int hiB = (lane >= 16) ? 16 : 0;

  const _Float16* Ar0 = A + (size_t)((tm << 5) + l15) * K;
  const _Float16* Ar1 = Ar0 + (size_t)16 * K;
  const _Float16* Br0 = B + (size_t)((tn << 6) + l15) * K;
  const _Float16* Br1 = Br0 + (size_t)16 * K;
  const _Float16* Br2 = Br0 + (size_t)32 * K;
  const _Float16* Br3 = Br0 + (size_t)48 * K;

  v16h a0 = ld16(Ar0 + hiA,  Ar0 + hiA + 8);
  v16h a1 = ld16(Ar1 + hiA,  Ar1 + hiA + 8);
  v16h b0 = ld16(Br0 + hiB,  Br0 + hiB + 8);
  v16h b1 = ld16(Br1 + hiB,  Br1 + hiB + 8);
  v16h b2 = ld16(Br2 + hiB,  Br2 + hiB + 8);
  v16h b3 = ld16(Br3 + hiB,  Br3 + hiB + 8);

  v8f c00 = {}, c01 = {}, c02 = {}, c03 = {};
  v8f c10 = {}, c11 = {}, c12 = {}, c13 = {};
  for (int kb = 0; kb < K; kb += 32) {
    const int kn = (kb + 32 < K) ? (kb + 32) : 0;   // uniform clamp
    // warm caches two K-steps ahead (global_prefetch_b8; speculative)
    __builtin_prefetch(Ar0 + kb + 64, 0, 3);
    __builtin_prefetch(Br0 + kb + 64, 0, 3);
    // issue next step's loads before consuming the current fragments
    v16h an0 = ld16(Ar0 + kn + hiA, Ar0 + kn + hiA + 8);
    v16h an1 = ld16(Ar1 + kn + hiA, Ar1 + kn + hiA + 8);
    v16h bn0 = ld16(Br0 + kn + hiB, Br0 + kn + hiB + 8);
    v16h bn1 = ld16(Br1 + kn + hiB, Br1 + kn + hiB + 8);
    v16h bn2 = ld16(Br2 + kn + hiB, Br2 + kn + hiB + 8);
    v16h bn3 = ld16(Br3 + kn + hiB, Br3 + kn + hiB + 8);
    c00 = __builtin_amdgcn_wmma_f32_16x16x32_f16(false, a0, false, b0, (short)0, c00, false, false);
    c10 = __builtin_amdgcn_wmma_f32_16x16x32_f16(false, a1, false, b0, (short)0, c10, false, false);
    c01 = __builtin_amdgcn_wmma_f32_16x16x32_f16(false, a0, false, b1, (short)0, c01, false, false);
    c11 = __builtin_amdgcn_wmma_f32_16x16x32_f16(false, a1, false, b1, (short)0, c11, false, false);
    c02 = __builtin_amdgcn_wmma_f32_16x16x32_f16(false, a0, false, b2, (short)0, c02, false, false);
    c12 = __builtin_amdgcn_wmma_f32_16x16x32_f16(false, a1, false, b2, (short)0, c12, false, false);
    c03 = __builtin_amdgcn_wmma_f32_16x16x32_f16(false, a0, false, b3, (short)0, c03, false, false);
    c13 = __builtin_amdgcn_wmma_f32_16x16x32_f16(false, a1, false, b3, (short)0, c13, false, false);
    a0 = an0; a1 = an1; b0 = bn0; b1 = bn1; b2 = bn2; b3 = bn3;
  }

  // C/D layout: VGPR v -> M = base + v (+8 for lanes 16-31); N = lane&15.
  const int hsel = (lane >= 16) ? 8 : 0;
#pragma unroll
  for (int part = 0; part < 2; ++part) {
    const int rbase = (tm << 5) + part * 16 + hsel;
    v8f t0 = part ? c10 : c00, t1 = part ? c11 : c01;
    v8f t2 = part ? c12 : c02, t3 = part ? c13 : c03;
#pragma unroll
    for (int v = 0; v < 8; ++v) {
      float* crow = C + (size_t)(rbase + v) * Nout + (tn << 6) + l15;
      float vals[4] = { t0[v], t1[v], t2[v], t3[v] };
#pragma unroll
      for (int nn = 0; nn < 4; ++nn) {
        float x = vals[nn];
        if (bias) x += bias[(tn << 6) + nn * 16 + l15];
        if (relu) x = fmaxf(x, 0.f);
        crow[nn * 16] = x;
      }
    }
  }
}

// --------------------------- f32 -> f16 convert ----------------------------
__global__ void f2h_kernel(const float4* __restrict__ src, v4h* __restrict__ dst,
                           size_t n4) {
  const size_t stride = (size_t)gridDim.x * blockDim.x;
  for (size_t i = (size_t)blockIdx.x * blockDim.x + threadIdx.x; i < n4; i += stride) {
    float4 v = src[i];
    v4h h;
    h[0] = (_Float16)v.x; h[1] = (_Float16)v.y;
    h[2] = (_Float16)v.z; h[3] = (_Float16)v.w;
    dst[i] = h;
  }
}

// --------------------------- banded softmax --------------------------------
// One wave per row i: logits over j in [i-ap+1, i+ap-1] (<=79), shuffle-
// reduced softmax.  Writes full [n,n] (video; zeros pre-filled) or banded
// [n,bstride] (audio).
__global__ void band_softmax(const float* __restrict__ Q, const float* __restrict__ Km,
                             float* __restrict__ outFull, float* __restrict__ outBand,
                             int n, int d, int ap, int bstride, float scale) {
  const int lane = threadIdx.x & 31;
  const int row  = blockIdx.x * (blockDim.x >> 5) + (threadIdx.x >> 5);
  const int cnt  = 2 * ap - 1;
  const float NEGINF = -__builtin_inff();

  const float4* q4 = (const float4*)(Q + (size_t)row * d);
  float lg[3];
  float lmax = NEGINF;
#pragma unroll
  for (int s = 0; s < 3; ++s) {
    const int t = lane + 32 * s;
    const int j = row - (ap - 1) + t;
    float v = NEGINF;
    if (t < cnt && j >= 0 && j < n) {
      const float4* k4 = (const float4*)(Km + (size_t)j * d);
      float acc = 0.f;
      for (int k = 0; k < (d >> 2); ++k) {
        float4 a = q4[k], b = k4[k];
        acc += a.x * b.x + a.y * b.y + a.z * b.z + a.w * b.w;
      }
      v = acc * scale;
    }
    lg[s] = v;
    lmax  = fmaxf(lmax, v);
  }
  lmax = wave_max(lmax);

  float e[3];
  float lsum = 0.f;
#pragma unroll
  for (int s = 0; s < 3; ++s) {
    e[s] = (lg[s] == NEGINF) ? 0.f : __expf(lg[s] - lmax);
    lsum += e[s];
  }
  lsum = wave_sum(lsum);
  const float inv = 1.f / lsum;

#pragma unroll
  for (int s = 0; s < 3; ++s) {
    const int t = lane + 32 * s;
    const int j = row - (ap - 1) + t;
    if (t < cnt && j >= 0 && j < n) {
      if (outFull) outFull[(size_t)row * n + j]       = e[s] * inv;
      else         outBand[(size_t)row * bstride + t] = e[s] * inv;
    }
  }
}

// y[i,:] = sum_j att[j,i] * V[j,:]   (transposed banded apply)
__global__ void band_applyT(const float* __restrict__ W, const float* __restrict__ V,
                            float* __restrict__ Y, int n, int d, int ap,
                            int isFull, int bstride) {
  const int i = blockIdx.x;
  const int half = ap - 1;
  int j0 = i - half; if (j0 < 0) j0 = 0;
  int j1 = i + half; if (j1 > n - 1) j1 = n - 1;
  const int nc = d / blockDim.x;             // <= 4 for all launches
  float acc[4] = {0.f, 0.f, 0.f, 0.f};
  for (int j = j0; j <= j1; ++j) {
    const float w = isFull ? W[(size_t)j * n + i]
                           : W[(size_t)j * bstride + (i - j + half)];
    for (int c = 0; c < nc; ++c) {
      const int dd = threadIdx.x + c * blockDim.x;
      acc[c] += w * V[(size_t)j * d + dd];
    }
  }
  for (int c = 0; c < nc; ++c)
    Y[(size_t)i * d + threadIdx.x + c * blockDim.x] = acc[c];
}

// --------------------------- layer norm ------------------------------------
__global__ void ln_kernel(const float* __restrict__ X, const float* __restrict__ R,
                          const float* __restrict__ g, const float* __restrict__ b,
                          float* __restrict__ O, int d) {
  const int i = blockIdx.x;
  __shared__ float s1[8], s2[8];
  float sum = 0.f, sq = 0.f;
  for (int k = threadIdx.x; k < d; k += blockDim.x) {
    float v = X[(size_t)i * d + k] + (R ? R[(size_t)i * d + k] : 0.f);
    sum += v; sq += v * v;
  }
  sum = wave_sum(sum); sq = wave_sum(sq);
  const int wid = threadIdx.x >> 5, lane = threadIdx.x & 31;
  const int nw = blockDim.x >> 5;
  if (lane == 0) { s1[wid] = sum; s2[wid] = sq; }
  __syncthreads();
  if (wid == 0) {
    float a = (lane < nw) ? s1[lane] : 0.f;
    float c = (lane < nw) ? s2[lane] : 0.f;
    a = wave_sum(a); c = wave_sum(c);
    if (lane == 0) { s1[0] = a; s2[0] = c; }
  }
  __syncthreads();
  const float mu   = s1[0] / d;
  const float var  = s2[0] / d - mu * mu;
  const float rstd = rsqrtf(var + 1e-6f);
  for (int k = threadIdx.x; k < d; k += blockDim.x) {
    float v = X[(size_t)i * d + k] + (R ? R[(size_t)i * d + k] : 0.f);
    O[(size_t)i * d + k] = (v - mu) * rstd * g[k] + b[k];
  }
}

// ----------------------------- small utils ---------------------------------
__global__ void zero_kernel(float* __restrict__ p, size_t count) {
  const size_t stride = (size_t)gridDim.x * blockDim.x * 4;
  for (size_t i = ((size_t)blockIdx.x * blockDim.x + threadIdx.x) * 4;
       i + 3 < count; i += stride) {
    *(float4*)(p + i) = make_float4(0.f, 0.f, 0.f, 0.f);
  }
}

// concat directly to f16 (only the MLP GEMM consumes it)
__global__ void concat_h_kernel(const float* __restrict__ a, const float* __restrict__ b,
                                _Float16* __restrict__ o, int n, int da, int db) {
  const int dt = da + db;
  const size_t total = (size_t)n * dt;
  for (size_t idx = (size_t)blockIdx.x * blockDim.x + threadIdx.x; idx < total;
       idx += (size_t)gridDim.x * blockDim.x) {
    const int row = (int)(idx / dt);
    const int c   = (int)(idx % dt);
    const float v = (c < da) ? a[(size_t)row * da + c] : b[(size_t)row * db + (c - da)];
    o[idx] = (_Float16)v;
  }
}

// out[i] = sigmoid(H[i,:] . w + b) ; one wave per row.
__global__ void head_kernel(const float* __restrict__ H, const float* __restrict__ w,
                            const float* __restrict__ b, float* __restrict__ out,
                            int n, int d) {
  const int i = blockIdx.x * (blockDim.x >> 5) + (threadIdx.x >> 5);
  const int lane = threadIdx.x & 31;
  float acc = 0.f;
  for (int k = lane; k < d; k += 32) acc += H[(size_t)i * d + k] * w[k];
  acc = wave_sum(acc);
  if (lane == 0) out[i] = 1.f / (1.f + __expf(-(acc + b[0])));
}

// ------------------------------- launch ------------------------------------
static inline void f2h(const float* s, _Float16* d, size_t count, hipStream_t st) {
  f2h_kernel<<<1024, 256, 0, st>>>((const float4*)s, (v4h*)d, count / 4);
}

extern "C" void kernel_launch(void* const* d_in, const int* in_sizes, int n_in,
                              void* d_out, int out_size, void* d_ws, size_t ws_size,
                              hipStream_t stream) {
  (void)in_sizes; (void)n_in; (void)out_size; (void)ws_size;
  const float* x     = (const float*)d_in[0];
  const float* audio = (const float*)d_in[1];
  const float* Wk    = (const float*)d_in[2];
  const float* Wq    = (const float*)d_in[3];
  const float* Wv    = (const float*)d_in[4];
  const float* Wo    = (const float*)d_in[5];
  const float* aWk   = (const float*)d_in[6];
  const float* aWq   = (const float*)d_in[7];
  const float* aWv   = (const float*)d_in[8];
  const float* aWo   = (const float*)d_in[9];
  const float* ka_w  = (const float*)d_in[10];
  const float* ka_b  = (const float*)d_in[11];
  const float* kd_w  = (const float*)d_in[12];
  const float* kd_b  = (const float*)d_in[13];
  const float* ln_y_g  = (const float*)d_in[14];
  const float* ln_y_b  = (const float*)d_in[15];
  const float* ln_ka_g = (const float*)d_in[16];
  const float* ln_ka_b = (const float*)d_in[17];
  const float* ln_a_g  = (const float*)d_in[18];
  const float* ln_a_b  = (const float*)d_in[19];
  const float* ln_a2_g = (const float*)d_in[20];
  const float* ln_a2_b = (const float*)d_in[21];

  float* out = (float*)d_out;            // [4096]
  float* att = out + N_SEQ;              // [4096*4096] (output #2)

  // ---- workspace layout (f32 region first, then 16B-aligned f16 region) ----
  float* ws = (float*)d_ws;
  size_t off = 0;
  float* audln = ws + off; off += (size_t)N_SEQ * A_DIM;
  float* Qa    = ws + off; off += (size_t)N_SEQ * A_DIM;
  float* Ka    = ws + off; off += (size_t)N_SEQ * A_DIM;
  float* Va    = ws + off; off += (size_t)N_SEQ * A_DIM;
  float* attA  = ws + off; off += (size_t)N_SEQ * 80;
  float* ya    = ws + off; off += (size_t)N_SEQ * A_DIM;
  float* Qv    = ws + off; off += (size_t)N_SEQ * M_DIM;
  float* Kv    = ws + off; off += (size_t)N_SEQ * M_DIM;
  float* Vv    = ws + off; off += (size_t)N_SEQ * M_DIM;

  _Float16* hbase = (_Float16*)(ws + off);
  size_t ho = 0;
  _Float16* x_h    = hbase + ho; ho += (size_t)N_SEQ * M_DIM;
  _Float16* Wk_h   = hbase + ho; ho += (size_t)M_DIM * M_DIM;
  _Float16* Wq_h   = hbase + ho; ho += (size_t)M_DIM * M_DIM;
  _Float16* Wv_h   = hbase + ho; ho += (size_t)M_DIM * M_DIM;
  _Float16* Wo_h   = hbase + ho; ho += (size_t)M_DIM * M_DIM;
  _Float16* aWk_h  = hbase + ho; ho += (size_t)A_DIM * A_DIM;
  _Float16* aWq_h  = hbase + ho; ho += (size_t)A_DIM * A_DIM;
  _Float16* aWv_h  = hbase + ho; ho += (size_t)A_DIM * A_DIM;
  _Float16* aWo_h  = hbase + ho; ho += (size_t)A_DIM * A_DIM;
  _Float16* kaw_h  = hbase + ho; ho += (size_t)M_DIM * (M_DIM + A_DIM);
  _Float16* audln_h= hbase + ho; ho += (size_t)N_SEQ * A_DIM;
  _Float16* ya_h   = hbase + ho; ho += (size_t)N_SEQ * A_DIM;
  _Float16* hcat_h = hbase + ho; ho += (size_t)N_SEQ * (M_DIM + A_DIM);
  _Float16* yv_h   = x_h;   // x_h dead after the three projections

  // f32 buffer reuse (each source fully consumed before overwrite)
  float* yv   = Qv;   // y_pre   (Qv consumed by band_softmax)
  float* yWo  = Kv;   // y@Wo^T  (Kv consumed by band_softmax)
  float* yln  = Vv;   // LN(y+x) (Vv consumed by band_applyT)
  float* yaWo = Qa;
  float* aln2 = Ka;
  float* h    = Qv;   // MLP hidden (yv consumed by yWo GEMM)
  float* hln  = Kv;   // LN(h)      (yWo consumed by yln)

  // ---- video att output: exact zeros outside the band ----
  zero_kernel<<<4096, 256, 0, stream>>>(att, (size_t)N_SEQ * N_SEQ);

  // ---- precision converts (bandwidth-bound, sub-us each) ----
  f2h(x,    x_h,   (size_t)N_SEQ * M_DIM, stream);
  f2h(Wk,   Wk_h,  (size_t)M_DIM * M_DIM, stream);
  f2h(Wq,   Wq_h,  (size_t)M_DIM * M_DIM, stream);
  f2h(Wv,   Wv_h,  (size_t)M_DIM * M_DIM, stream);
  f2h(Wo,   Wo_h,  (size_t)M_DIM * M_DIM, stream);
  f2h(aWk,  aWk_h, (size_t)A_DIM * A_DIM, stream);
  f2h(aWq,  aWq_h, (size_t)A_DIM * A_DIM, stream);
  f2h(aWv,  aWv_h, (size_t)A_DIM * A_DIM, stream);
  f2h(aWo,  aWo_h, (size_t)A_DIM * A_DIM, stream);
  f2h(ka_w, kaw_h, (size_t)M_DIM * (M_DIM + A_DIM), stream);

  // ---- audio input LN (ln_a2) ----
  ln_kernel<<<N_SEQ, 128, 0, stream>>>(audio, nullptr, ln_a2_g, ln_a2_b, audln, A_DIM);
  f2h(audln, audln_h, (size_t)N_SEQ * A_DIM, stream);

  // ---- video projections: (4096/32)*(1024/64)=2048 tiles, 4 waves/block ----
  wmma_gemm_h<<<512, 128, 0, stream>>>(x_h, Wk_h, nullptr, Kv, N_SEQ, M_DIM, M_DIM, 0);
  wmma_gemm_h<<<512, 128, 0, stream>>>(x_h, Wq_h, nullptr, Qv, N_SEQ, M_DIM, M_DIM, 0);
  wmma_gemm_h<<<512, 128, 0, stream>>>(x_h, Wv_h, nullptr, Vv, N_SEQ, M_DIM, M_DIM, 0);

  // ---- video banded softmax (aperture 20), writes full att matrix ----
  band_softmax<<<512, 256, 0, stream>>>(Qv, Kv, att, nullptr, N_SEQ, M_DIM, 20, 0, 0.06f);

  // ---- y = att.T @ V (banded), then @ Wo^T, residual + LN ----
  band_applyT<<<N_SEQ, 256, 0, stream>>>(att, Vv, yv, N_SEQ, M_DIM, 20, 1, 0);
  f2h(yv, yv_h, (size_t)N_SEQ * M_DIM, stream);       // x_h dead from here on
  wmma_gemm_h<<<512, 128, 0, stream>>>(yv_h, Wo_h, nullptr, yWo, N_SEQ, M_DIM, M_DIM, 0);
  ln_kernel<<<N_SEQ, 256, 0, stream>>>(yWo, x, ln_y_g, ln_y_b, yln, M_DIM);

  // ---- audio attention (aperture 40): (4096/32)*(128/64)=256 tiles ----
  wmma_gemm_h<<<64, 128, 0, stream>>>(audln_h, aWk_h, nullptr, Ka, N_SEQ, A_DIM, A_DIM, 0);
  wmma_gemm_h<<<64, 128, 0, stream>>>(audln_h, aWq_h, nullptr, Qa, N_SEQ, A_DIM, A_DIM, 0);
  wmma_gemm_h<<<64, 128, 0, stream>>>(audln_h, aWv_h, nullptr, Va, N_SEQ, A_DIM, A_DIM, 0);
  band_softmax<<<512, 256, 0, stream>>>(Qa, Ka, nullptr, attA, N_SEQ, A_DIM, 40, 80, 0.06f);
  band_applyT<<<N_SEQ, 128, 0, stream>>>(attA, Va, ya, N_SEQ, A_DIM, 40, 0, 80);
  f2h(ya, ya_h, (size_t)N_SEQ * A_DIM, stream);
  wmma_gemm_h<<<64, 128, 0, stream>>>(ya_h, aWo_h, nullptr, yaWo, N_SEQ, A_DIM, A_DIM, 0);
  ln_kernel<<<N_SEQ, 128, 0, stream>>>(yaWo, audln, ln_a_g, ln_a_b, aln2, A_DIM);

  // ---- MLP head: concat(f16) -> relu GEMM (K=1152) -> LN -> sigmoid dot ----
  concat_h_kernel<<<2304, 256, 0, stream>>>(yln, aln2, hcat_h, N_SEQ, M_DIM, A_DIM);
  wmma_gemm_h<<<512, 128, 0, stream>>>(hcat_h, kaw_h, ka_b, h, N_SEQ, M_DIM, M_DIM + A_DIM, 1);
  ln_kernel<<<N_SEQ, 256, 0, stream>>>(h, nullptr, ln_ka_g, ln_ka_b, hln, M_DIM);
  head_kernel<<<512, 256, 0, stream>>>(hln, kd_w, kd_b, out, N_SEQ, M_DIM);
}
